// GCN_3307124818738
// MI455X (gfx1250) — matrix operand order repaired and checked by reading, not compiled, not run on hardware
//
#include <hip/hip_runtime.h>
#include <hip/hip_bf16.h>

// ---------------------------------------------------------------------------
// 2-layer GCN for MI455X (gfx1250, wave32).
//   Layer1 GEMM: v_wmma_f32_16x16x32_bf16, W1 staged in LDS (ds_load_b128),
//                2 M-tiles per wave for B-fragment reuse.
//   Layer2 GEMM: v_wmma_f32_16x16x4_f32 (full fp32 precision, small)
//   Edge aggregation: hardware global_atomic_add_f32; H1/A1 (51 MB each)
//   fit in the 192 MB L2 so gather/scatter stays on-chip.
// ---------------------------------------------------------------------------

typedef __bf16 bf16_t;
typedef bf16_t v16bf __attribute__((ext_vector_type(16)));
typedef float  v8f   __attribute__((ext_vector_type(8)));
typedef float  v2f   __attribute__((ext_vector_type(2)));

#define GCN_N       100000
#define GCN_INF     512
#define GCN_HID     128
#define GCN_CLS     32
#define W1_ELEMS    (GCN_INF * GCN_HID)      // 65536 bf16 = 128 KB

// ---- workspace layout (bytes, 256-aligned) --------------------------------
#define OFF_DINV    0u            // 100000 f32            (400000 B)
#define OFF_WB      400128u       // 512*128 bf16 swizzled (131072 B)
#define OFF_W2S     531200u       // 4096 f32 swizzled
#define OFF_H1      547584u       // 100000*128 f32        (51200000 B)
#define OFF_A1      51747584u     // 100000*128 f32
#define OFF_H2      102947584u    // 100000*32 f32

// ---------------------------------------------------------------------------
__global__ void k_deg_init(float* __restrict__ deg, int n) {
    int i = blockIdx.x * blockDim.x + threadIdx.x;
    if (i < n) deg[i] = 1.0f;                       // self-loop contributes 1
}

__global__ void k_deg_edges(const long long* __restrict__ dst,
                            float* __restrict__ deg, int nE) {
    int e = blockIdx.x * blockDim.x + threadIdx.x;
    if (e < nE) unsafeAtomicAdd(&deg[dst[e]], 1.0f);
}

__global__ void k_dinv(float* __restrict__ deg, int n) {
    int i = blockIdx.x * blockDim.x + threadIdx.x;
    if (i < n) {
        float d = deg[i];
        deg[i] = (d > 0.0f) ? rsqrtf(d) : 0.0f;
    }
}

// ---- pre-swizzle W1 (f32 -> bf16) into WMMA B-fragment lane layout --------
// slot t = ((kc*8 + nt)*32 + lane)*16 + i ; lane: n=lane&15, khalf=lane>>4
// element i: krow = kc*32 + (i>>3)*16 + khalf*8 + (i&7) ; col = nt*16+n
__global__ void k_wswz1(const float* __restrict__ W1, bf16_t* __restrict__ Wb) {
    int t = blockIdx.x * blockDim.x + threadIdx.x;          // 65536 threads
    int i    = t & 15;
    int lane = (t >> 4) & 31;
    int ntk  = t >> 9;
    int nt   = ntk & 7;
    int kc   = ntk >> 3;
    int n    = lane & 15;
    int khalf = lane >> 4;
    int krow = kc * 32 + ((i >> 3) << 4) + khalf * 8 + (i & 7);
    int col  = nt * 16 + n;
    Wb[t] = (bf16_t)W1[krow * GCN_HID + col];
}

// ---- pre-swizzle W2 into f32-WMMA B-fragment layout -----------------------
// slot t = (((kq*2)+nt)*32 + lane)*2 + p ; k = kq*4 + 2*khalf + p
__global__ void k_wswz2(const float* __restrict__ W2, float* __restrict__ W2s) {
    int t = blockIdx.x * blockDim.x + threadIdx.x;          // 4096 threads
    int p    = t & 1;
    int lane = (t >> 1) & 31;
    int ntk  = t >> 6;
    int nt   = ntk & 1;
    int kq   = ntk >> 1;
    int n    = lane & 15;
    int khalf = lane >> 4;
    int k = kq * 4 + 2 * khalf + p;
    W2s[t] = W2[k * GCN_CLS + nt * 16 + n];
}

// ---- GEMM1: H1 = x @ W1  (bf16 WMMA, fp32 accum) --------------------------
// Block = 8 waves; whole swizzled W1 staged in LDS (128 KB).
// One wave -> 32 rows (two 16-row M-tiles) x 128 cols: each LDS B fragment
// feeds two WMMAs.
__global__ __launch_bounds__(256) void k_gemm1(const float* __restrict__ x,
                                               const bf16_t* __restrict__ Wb,
                                               float* __restrict__ H1) {
    extern __shared__ char smem[];
    bf16_t* sW = (bf16_t*)smem;                          // 65536 bf16

    const int tid  = threadIdx.x;
    const int lane = tid & 31;
    const int wave = tid >> 5;

    // cooperative stage of swizzled W1 into LDS (8192 x uint4)
    {
        const uint4* g = (const uint4*)Wb;
        uint4* l = (uint4*)sW;
        #pragma unroll
        for (int i = 0; i < W1_ELEMS / 8 / 256; ++i)      // 32 iters
            l[tid + i * 256] = g[tid + i * 256];
    }
    __syncthreads();

    const int row0 = blockIdx.x * 256 + wave * 32;        // 2 M-tiles per wave
    const int m    = lane & 15;
    const int half = lane >> 4;

    int r0 = row0 + m;        if (r0 >= GCN_N) r0 = GCN_N - 1;
    int r1 = row0 + 16 + m;   if (r1 >= GCN_N) r1 = GCN_N - 1;
    const float* __restrict__ xrow0 = x + (size_t)r0 * GCN_INF;
    const float* __restrict__ xrow1 = x + (size_t)r1 * GCN_INF;

    v8f acc0[8] = {};
    v8f acc1[8] = {};

    for (int kc = 0; kc < GCN_INF / 32; ++kc) {
        const float* pa0 = xrow0 + kc * 32 + half * 8;
        const float* pa1 = xrow1 + kc * 32 + half * 8;
        if (kc + 1 < GCN_INF / 32) {
            __builtin_prefetch(pa0 + 32, 0, 3);
            __builtin_prefetch(pa1 + 32, 0, 3);
        }
        // A fragment: lane needs K[half*8 .. +7] and K[16+half*8 .. +7]
        float4 s0 = *(const float4*)(pa0 + 0);
        float4 s1 = *(const float4*)(pa0 + 4);
        float4 s2 = *(const float4*)(pa0 + 16);
        float4 s3 = *(const float4*)(pa0 + 20);
        float4 u0 = *(const float4*)(pa1 + 0);
        float4 u1 = *(const float4*)(pa1 + 4);
        float4 u2 = *(const float4*)(pa1 + 16);
        float4 u3 = *(const float4*)(pa1 + 20);

        v16bf a0, a1;
        a0[0]=(bf16_t)s0.x; a0[1]=(bf16_t)s0.y; a0[2]=(bf16_t)s0.z; a0[3]=(bf16_t)s0.w;
        a0[4]=(bf16_t)s1.x; a0[5]=(bf16_t)s1.y; a0[6]=(bf16_t)s1.z; a0[7]=(bf16_t)s1.w;
        a0[8]=(bf16_t)s2.x; a0[9]=(bf16_t)s2.y; a0[10]=(bf16_t)s2.z; a0[11]=(bf16_t)s2.w;
        a0[12]=(bf16_t)s3.x; a0[13]=(bf16_t)s3.y; a0[14]=(bf16_t)s3.z; a0[15]=(bf16_t)s3.w;
        a1[0]=(bf16_t)u0.x; a1[1]=(bf16_t)u0.y; a1[2]=(bf16_t)u0.z; a1[3]=(bf16_t)u0.w;
        a1[4]=(bf16_t)u1.x; a1[5]=(bf16_t)u1.y; a1[6]=(bf16_t)u1.z; a1[7]=(bf16_t)u1.w;
        a1[8]=(bf16_t)u2.x; a1[9]=(bf16_t)u2.y; a1[10]=(bf16_t)u2.z; a1[11]=(bf16_t)u2.w;
        a1[12]=(bf16_t)u3.x; a1[13]=(bf16_t)u3.y; a1[14]=(bf16_t)u3.z; a1[15]=(bf16_t)u3.w;

        #pragma unroll
        for (int nt = 0; nt < 8; ++nt) {
            const v16bf b = *(const v16bf*)(sW +
                (((kc * 8 + nt) * 32 + lane) << 4));      // ds_load_b128 x2
            acc0[nt] = __builtin_amdgcn_wmma_f32_16x16x32_bf16(
                false, a0, false, b, (short)0, acc0[nt], false, false);
            acc1[nt] = __builtin_amdgcn_wmma_f32_16x16x32_bf16(
                false, a1, false, b, (short)0, acc1[nt], false, false);
        }
    }

    const int n = lane & 15;
    #pragma unroll
    for (int nt = 0; nt < 8; ++nt) {
        #pragma unroll
        for (int rr = 0; rr < 8; ++rr) {
            int M0 = row0 + rr + 8 * half;                // C/D: M = r + 8*half
            int M1 = M0 + 16;
            if (M0 < GCN_N)
                H1[(size_t)M0 * GCN_HID + nt * 16 + n] = acc0[nt][rr];
            if (M1 < GCN_N)
                H1[(size_t)M1 * GCN_HID + nt * 16 + n] = acc1[nt][rr];
        }
    }
}

// ---- A1 = b1 + H1 * dinv^2  (self-loop message + bias) --------------------
__global__ void k_init_a1(const float* __restrict__ H1,
                          const float* __restrict__ dinv,
                          const float* __restrict__ b1,
                          float* __restrict__ A1) {
    size_t i = (size_t)blockIdx.x * blockDim.x + threadIdx.x;
    if (i >= (size_t)GCN_N * GCN_HID) return;
    int node = (int)(i >> 7);
    int f    = (int)(i & 127);
    float w = dinv[node];
    A1[i] = b1[f] + H1[i] * w * w;
}

// ---- layer-1 edge scatter: one wave per edge, float4 per lane -------------
__global__ __launch_bounds__(256) void k_agg1(const long long* __restrict__ src,
                                              const long long* __restrict__ dst,
                                              const float* __restrict__ dinv,
                                              const float* __restrict__ H1,
                                              float* __restrict__ A1, int nE) {
    int e    = (int)(((size_t)blockIdx.x * blockDim.x + threadIdx.x) >> 5);
    int lane = threadIdx.x & 31;
    if (e >= nE) return;
    long long s = src[e], d = dst[e];
    float w = dinv[s] * dinv[d];
    const float4 v = *(const float4*)(H1 + (size_t)s * GCN_HID + lane * 4);
    float* o = A1 + (size_t)d * GCN_HID + lane * 4;
    unsafeAtomicAdd(o + 0, v.x * w);
    unsafeAtomicAdd(o + 1, v.y * w);
    unsafeAtomicAdd(o + 2, v.z * w);
    unsafeAtomicAdd(o + 3, v.w * w);
}

__global__ void k_relu(float* __restrict__ A1) {
    size_t i = (size_t)blockIdx.x * blockDim.x + threadIdx.x;
    if (i >= (size_t)GCN_N * GCN_HID) return;
    float v = A1[i];
    A1[i] = v > 0.0f ? v : 0.0f;
}

// ---- GEMM2: H2 = A1 @ W2  (fp32 WMMA 16x16x4) -----------------------------
__global__ __launch_bounds__(256) void k_gemm2(const float* __restrict__ A1,
                                               const float* __restrict__ W2s,
                                               float* __restrict__ H2) {
    const int lane = threadIdx.x & 31;
    const int wave = threadIdx.x >> 5;
    const int row0 = blockIdx.x * 128 + wave * 16;
    const int m    = lane & 15;
    const int half = lane >> 4;

    int r = row0 + m;
    if (r >= GCN_N) r = GCN_N - 1;
    const float* __restrict__ arow = A1 + (size_t)r * GCN_HID;

    v8f acc[2] = {};

    for (int kq = 0; kq < GCN_HID / 4; ++kq) {
        // f32 A 16x4 layout: VGPR0=K(2*half), VGPR1=K(2*half+1)
        v2f a = *(const v2f*)(arow + kq * 4 + 2 * half);
        #pragma unroll
        for (int nt = 0; nt < 2; ++nt) {
            v2f b = *(const v2f*)(W2s + ((size_t)((kq * 2 + nt) * 32 + lane) << 1));
            acc[nt] = __builtin_amdgcn_wmma_f32_16x16x4_f32(
                false, a, false, b, (short)0, acc[nt], false, false);
        }
    }

    const int n = lane & 15;
    #pragma unroll
    for (int nt = 0; nt < 2; ++nt) {
        #pragma unroll
        for (int rr = 0; rr < 8; ++rr) {
            int Mrow = row0 + rr + 8 * half;
            if (Mrow < GCN_N)
                H2[(size_t)Mrow * GCN_CLS + nt * 16 + n] = acc[nt][rr];
        }
    }
}

// ---- out = b2 + H2 * dinv^2 ----------------------------------------------
__global__ void k_init_out(const float* __restrict__ H2,
                           const float* __restrict__ dinv,
                           const float* __restrict__ b2,
                           float* __restrict__ out) {
    size_t i = (size_t)blockIdx.x * blockDim.x + threadIdx.x;
    if (i >= (size_t)GCN_N * GCN_CLS) return;
    int node = (int)(i >> 5);
    int f    = (int)(i & 31);
    float w = dinv[node];
    out[i] = b2[f] + H2[i] * w * w;
}

// ---- layer-2 edge scatter: one wave per edge, 1 float per lane ------------
__global__ __launch_bounds__(256) void k_agg2(const long long* __restrict__ src,
                                              const long long* __restrict__ dst,
                                              const float* __restrict__ dinv,
                                              const float* __restrict__ H2,
                                              float* __restrict__ out, int nE) {
    int e    = (int)(((size_t)blockIdx.x * blockDim.x + threadIdx.x) >> 5);
    int lane = threadIdx.x & 31;
    if (e >= nE) return;
    long long s = src[e], d = dst[e];
    float w = dinv[s] * dinv[d];
    float v = H2[(size_t)s * GCN_CLS + lane] * w;
    unsafeAtomicAdd(out + (size_t)d * GCN_CLS + lane, v);
}

// ---------------------------------------------------------------------------
extern "C" void kernel_launch(void* const* d_in, const int* in_sizes, int n_in,
                              void* d_out, int out_size, void* d_ws, size_t ws_size,
                              hipStream_t stream) {
    const float*     x   = (const float*)d_in[0];
    const long long* ei  = (const long long*)d_in[1];     // int64 edge_index
    const float*     W1  = (const float*)d_in[2];
    const float*     b1  = (const float*)d_in[3];
    const float*     W2  = (const float*)d_in[4];
    const float*     b2  = (const float*)d_in[5];
    float*           out = (float*)d_out;

    const int nE = in_sizes[1] / 2;                       // 1,600,000
    const long long* src = ei;
    const long long* dst = ei + nE;

    char* ws = (char*)d_ws;
    float*  dinv = (float*)(ws + OFF_DINV);
    bf16_t* Wb   = (bf16_t*)(ws + OFF_WB);
    float*  W2s  = (float*)(ws + OFF_W2S);
    float*  H1   = (float*)(ws + OFF_H1);
    float*  A1   = (float*)(ws + OFF_A1);
    float*  H2   = (float*)(ws + OFF_H2);

    const int T = 256;
    // degrees & normalization
    k_deg_init<<<(GCN_N + T - 1) / T, T, 0, stream>>>(dinv, GCN_N);
    k_deg_edges<<<(nE + T - 1) / T, T, 0, stream>>>(dst, dinv, nE);
    k_dinv<<<(GCN_N + T - 1) / T, T, 0, stream>>>(dinv, GCN_N);
    // weight pre-swizzles
    k_wswz1<<<W1_ELEMS / T, T, 0, stream>>>(W1, Wb);
    k_wswz2<<<(GCN_HID / 4) * 2 * 32 * 2 / T, T, 0, stream>>>(W2, W2s);
    // layer 1  (dynamic LDS: whole swizzled W1 = 131072 B)
    k_gemm1<<<(GCN_N + 255) / 256, T, W1_ELEMS * sizeof(bf16_t), stream>>>(x, Wb, H1);
    k_init_a1<<<((size_t)GCN_N * GCN_HID + T - 1) / T, T, 0, stream>>>(H1, dinv, b1, A1);
    k_agg1<<<((size_t)nE * 32 + T - 1) / T, T, 0, stream>>>(src, dst, dinv, H1, A1, nE);
    k_relu<<<((size_t)GCN_N * GCN_HID + T - 1) / T, T, 0, stream>>>(A1);
    // layer 2
    k_gemm2<<<(GCN_N + 127) / 128, T, 0, stream>>>(A1, W2s, H2);
    k_init_out<<<((size_t)GCN_N * GCN_CLS + T - 1) / T, T, 0, stream>>>(H2, dinv, b2, out);
    k_agg2<<<((size_t)nE * 32 + T - 1) / T, T, 0, stream>>>(src, dst, dinv, H2, out, nE);
}
